// RecurrentDecoder_42494406426982
// MI455X (gfx1250) — matrix operand order repaired and checked by reading, not compiled, not run on hardware
//
#include <hip/hip_runtime.h>
#include <hip/hip_bf16.h>

#define C_ 15
#define H_ 256
#define G3_ 768
#define O_ 512
#define T_ 60
#define B_ 1024
#define PITCH 264        // h row pitch in halfs (256 + 8) -> rotates LDS banks per row
#define GPITCH 772       // gi row pitch in floats (768 + 4) -> 16B bank shift per row
#define ROWS 32          // batch rows per block
#define MAXTILE 47       // sum ceil(cnt_c/32) <= floor(1024/32) + 15 = 47

typedef __attribute__((ext_vector_type(16))) _Float16 v16h;
typedef __attribute__((ext_vector_type(8)))  _Float16 v8h;
typedef __attribute__((ext_vector_type(8)))  float    v8f;

// ---------------- one-time weight f32 -> f16 conversion ----------------
__global__ __launch_bounds__(256)
void wconv_kernel(const float* __restrict__ whh, const float* __restrict__ wc,
                  _Float16* __restrict__ whh16, _Float16* __restrict__ wc16) {
  const int NW1 = C_ * G3_ * H_;
  const int NW2 = C_ * O_ * H_;
  int i = blockIdx.x * 256 + threadIdx.x;
  if (i < NW1) {
    whh16[i] = (_Float16)whh[i];
  } else {
    i -= NW1;
    if (i < NW2) wc16[i] = (_Float16)wc[i];
  }
}

// ---------------- counting sort of batch indices by camera ----------------
__global__ __launch_bounds__(1024)
void sort_kernel(const int* __restrict__ cam, int* __restrict__ order,
                 int* __restrict__ tcam, int* __restrict__ tbase) {
  __shared__ int cnt[C_], cur[C_];
  int tid = threadIdx.x;
  if (tid < C_) { cnt[tid] = 0; cur[tid] = 0; }
  for (int i = tid; i < C_ * B_; i += 1024) order[i] = -1;   // padding sentinel
  __syncthreads();
  int c = cam[tid];
  atomicAdd(&cnt[c], 1);
  __syncthreads();
  int pos = atomicAdd(&cur[c], 1);
  order[c * B_ + pos] = tid;
  __syncthreads();
  if (tid == 0) {
    int k = 0;
    for (int cc = 0; cc < C_; ++cc) {
      int nt = (cnt[cc] + ROWS - 1) / ROWS;
      for (int j = 0; j < nt; ++j) { tcam[k] = cc; tbase[k] = cc * B_ + j * ROWS; ++k; }
    }
    for (; k < MAXTILE; ++k) tcam[k] = -1;
  }
}

// ---------------- gi = x . W_ih^T + b_ih  (one-time, f32) ----------------
__global__ __launch_bounds__(256)
void gi_kernel(const float* __restrict__ x, const int* __restrict__ cam,
               const float* __restrict__ W_ih, const float* __restrict__ b_ih,
               float* __restrict__ gi) {
  int idx = blockIdx.x * 256 + threadIdx.x;
  if (idx >= B_ * G3_) return;
  int b = idx / G3_, g = idx % G3_;
  int c = cam[b];
  const float* w  = W_ih + ((size_t)c * G3_ + g) * H_;
  const float* xv = x + (size_t)b * H_;
  float s = b_ih[c * G3_ + g];
  for (int k = 0; k < H_; k += 4) {
    float4 wv = *(const float4*)(w + k);
    float4 xw = *(const float4*)(xv + k);
    s += wv.x * xw.x + wv.y * xw.y + wv.z * xw.z + wv.w * xw.w;
  }
  gi[idx] = s;
}

// ---------------- A-operand load: h tile (16 x 256) from LDS -------------
// 16-bit A layout (16x32), contiguous k order: lane group g = lane/16 holds
// K = g*16..g*16+15 (matches B loads), row M = lane%16.
__device__ __forceinline__
void load_A(const _Float16* hb, int m0, int lmod, int lgrp, v16h* A) {
  const _Float16* hp = hb + (m0 + lmod) * PITCH;
  #pragma unroll
  for (int kt = 0; kt < 8; ++kt) {
    int kb = kt * 32 + lgrp * 16;
    v8h lo = *(const v8h*)(hp + kb);
    v8h hi = *(const v8h*)(hp + kb + 8);
    v16h a;
    #pragma unroll
    for (int e = 0; e < 8; ++e) { a[e] = lo[e]; a[e + 8] = hi[e]; }
    A[kt] = a;
  }
}

__device__ __forceinline__ float sigm(float v) { return 1.f / (1.f + __expf(-v)); }

// ---------------- main persistent GRU kernel -----------------------------
// 512 threads = 16 waves. Wave w: m-tile mt = w&1 (rows m0..m0+15), column
// slice ch = w>>1 (0..7). Per step: gate GEMM (2 h-col tiles x 3 gates,
// software-pipelined B loads), in-register GRU update, h write, A reload,
// output GEMM (4 o-tiles) + sigmoid + store. All weight addresses are
// computed inline from kernel-arg bases so loads stay global_load (not flat).
__global__ __launch_bounds__(512)
void gru_main(const float* __restrict__ gi, const float* __restrict__ b_hh,
              const float* __restrict__ bc, const _Float16* __restrict__ Wh16,
              const _Float16* __restrict__ Wc16, const int* __restrict__ order,
              const int* __restrict__ tcam, const int* __restrict__ tbase,
              float* __restrict__ out) {
  __shared__ int sidx[ROWS];
  __shared__ _Float16 hbuf[ROWS * PITCH];
  __shared__ float gbuf[ROWS * GPITCH];

  int c = tcam[blockIdx.x];
  if (c < 0) return;
  int base = tbase[blockIdx.x];

  int tid  = threadIdx.x;
  int lane = tid & 31;
  int wid  = tid >> 5;
  int lmod = lane & 15;   // N column / A row within tile
  int lgrp = lane >> 4;   // lane half-group
  int mt   = wid & 1;
  int ch   = wid >> 1;    // 0..7
  int m0   = mt * 16;

  if (tid < ROWS) sidx[tid] = order[base + tid];
  for (int i = tid; i < ROWS * PITCH; i += 512) hbuf[i] = (_Float16)0.f;
  __syncthreads();
  // stage time-invariant gi tile into LDS once; fold r/z gate biases in
  // (they are purely additive: sigmoid(gi + h.Wh + bh)).
  for (int i = tid; i < ROWS * G3_; i += 512) {
    int m = i / G3_, col = i - m * G3_;
    int brow = sidx[m];
    float v = (brow >= 0) ? gi[(size_t)brow * G3_ + col] : 0.f;
    if (col < 2 * H_) v += b_hh[c * G3_ + col];   // bh_r, bh_z folded here
    gbuf[m * GPITCH + col] = v;
  }

  const _Float16* Wh = Wh16 + (size_t)c * G3_ * H_;
  const _Float16* Wc = Wc16 + (size_t)c * O_ * H_;

  // per-row output base offsets (brow * O * T); -1 rows -> negative sentinel
  long long obase[8];
  #pragma unroll
  for (int r = 0; r < 8; ++r) {
    int brow = sidx[m0 + r + 8 * lgrp];
    obase[r] = (brow >= 0) ? (long long)brow * (O_ * T_) : -1;
  }
  __syncthreads();

  v16h A[8];
  load_A(hbuf, m0, lmod, lgrp, A);     // h0 = 0

  for (int t = 0; t < T_; ++t) {
    __syncthreads();   // all A-reg loads complete before anyone writes h

    // ---- gate GEMM + in-register GRU update (2 h-col tiles per wave) ----
    #pragma unroll 1
    for (int jj = 0; jj < 2; ++jj) {
      int j  = ch * 2 + jj;              // h-col tile 0..15
      int nR = j * 16 + lmod;
      // inline base-relative addresses -> global_load_b128
      const _Float16* wR = Wh + (size_t)nR * H_ + lgrp * 16;
      const _Float16* wZ = wR + (size_t)H_ * H_;        // +256 rows
      const _Float16* wN = wR + (size_t)2 * H_ * H_;    // +512 rows
      float bN0 = b_hh[c * G3_ + 2 * H_ + nR];          // bh_n (inside tanh*r)
      v8f aR, aZ, aN;
      #pragma unroll
      for (int r = 0; r < 8; ++r) { aR[r] = 0.f; aZ[r] = 0.f; aN[r] = bN0; }
      // software pipeline: prefetch next k-tile's B operands before consuming
      v16h bRa = *(const v16h*)(wR);
      v16h bZa = *(const v16h*)(wZ);
      v16h bNa = *(const v16h*)(wN);
      #pragma unroll
      for (int kt = 0; kt < 8; ++kt) {
        v16h bRb = bRa, bZb = bZa, bNb = bNa;
        if (kt < 7) {
          bRb = *(const v16h*)(wR + (kt + 1) * 32);
          bZb = *(const v16h*)(wZ + (kt + 1) * 32);
          bNb = *(const v16h*)(wN + (kt + 1) * 32);
        }
        aR = __builtin_amdgcn_wmma_f32_16x16x32_f16(false, A[kt], false, bRa, (short)0, aR, false, false);
        aZ = __builtin_amdgcn_wmma_f32_16x16x32_f16(false, A[kt], false, bZa, (short)0, aZ, false, false);
        aN = __builtin_amdgcn_wmma_f32_16x16x32_f16(false, A[kt], false, bNa, (short)0, aN, false, false);
        bRa = bRb; bZa = bZb; bNa = bNb;
      }
      int col = j * 16 + lmod;
      #pragma unroll
      for (int r = 0; r < 8; ++r) {    // C layout: M = r + 8*lgrp, N = lmod
        int m = m0 + r + 8 * lgrp;
        const float* gp = &gbuf[m * GPITCH + col];
        float hold = (float)hbuf[m * PITCH + col];
        float rr = sigm(gp[0] + aR[r]);          // bh_r already in gbuf
        float zz = sigm(gp[256] + aZ[r]);        // bh_z already in gbuf
        float nn = tanhf(gp[512] + rr * aN[r]);  // bh_n inside aN
        float hn = (1.f - zz) * nn + zz * hold;
        hbuf[m * PITCH + col] = (_Float16)hn;    // cell owned by this thread
      }
    }
    __syncthreads();   // h_new fully written

    load_A(hbuf, m0, lmod, lgrp, A);  // reload A = h_new (reused next step)

    // ---- output GEMM: out = sigmoid(h_new . Wc^T + bc), 4 o-tiles/wave ----
    #pragma unroll 1
    for (int oo = 0; oo < 4; ++oo) {
      int n = (ch * 4 + oo) * 16 + lmod;
      const _Float16* wO = Wc + (size_t)n * H_ + lgrp * 16;
      float bv = bc[c * O_ + n];
      v8f acc;
      #pragma unroll
      for (int r = 0; r < 8; ++r) acc[r] = bv;
      v16h bOa = *(const v16h*)(wO);
      #pragma unroll
      for (int kt = 0; kt < 8; ++kt) {
        v16h bOb = bOa;
        if (kt < 7) bOb = *(const v16h*)(wO + (kt + 1) * 32);
        acc = __builtin_amdgcn_wmma_f32_16x16x32_f16(false, A[kt], false, bOa, (short)0, acc, false, false);
        bOa = bOb;
      }
      long long ocol = (long long)(n * T_ + t);
      #pragma unroll
      for (int r = 0; r < 8; ++r) {
        if (obase[r] >= 0) out[obase[r] + ocol] = sigm(acc[r]);
      }
    }
  }
}

extern "C" void kernel_launch(void* const* d_in, const int* in_sizes, int n_in,
                              void* d_out, int out_size, void* d_ws, size_t ws_size,
                              hipStream_t stream) {
  const float* x    = (const float*)d_in[0];
  const int*   cam  = (const int*)d_in[1];
  const float* W_ih = (const float*)d_in[2];
  const float* W_hh = (const float*)d_in[3];
  const float* b_ih = (const float*)d_in[4];
  const float* b_hh = (const float*)d_in[5];
  const float* Wc   = (const float*)d_in[6];
  const float* bc   = (const float*)d_in[7];
  float* out = (float*)d_out;

  char* ws = (char*)d_ws;
  size_t off = 0;
  _Float16* Wh16 = (_Float16*)(ws + off); off += (size_t)C_ * G3_ * H_ * 2;
  _Float16* Wc16 = (_Float16*)(ws + off); off += (size_t)C_ * O_  * H_ * 2;
  float*    gi   = (float*)   (ws + off); off += (size_t)B_ * G3_ * 4;
  int*      order= (int*)     (ws + off); off += (size_t)C_ * B_  * 4;
  int*      tcam = (int*)     (ws + off); off += MAXTILE * 4;
  int*      tbase= (int*)     (ws + off); off += MAXTILE * 4;
  (void)ws_size; (void)in_sizes; (void)n_in; (void)out_size;

  int nconv = C_ * G3_ * H_ + C_ * O_ * H_;
  wconv_kernel<<<(nconv + 255) / 256, 256, 0, stream>>>(W_hh, Wc, Wh16, Wc16);
  sort_kernel<<<1, 1024, 0, stream>>>(cam, order, tcam, tbase);
  gi_kernel<<<(B_ * G3_ + 255) / 256, 256, 0, stream>>>(x, cam, W_ih, b_ih, gi);
  gru_main<<<MAXTILE, 512, 0, stream>>>(gi, b_hh, bc, Wh16, Wc16, order, tcam, tbase, out);
}
